// MyLSTM_82952998355377
// MI455X (gfx1250) — matrix operand order repaired and checked by reading, not compiled, not run on hardware
//
#include <hip/hip_runtime.h>
#include <hip/hip_bf16.h>
#include <stdint.h>

// ---------------------------------------------------------------------------
// Types for CDNA5 WMMA
// ---------------------------------------------------------------------------
typedef __attribute__((ext_vector_type(16))) __bf16 v16bf;
typedef __attribute__((ext_vector_type(8)))  float  v8f;
typedef __attribute__((ext_vector_type(4)))  unsigned int v4u;

union FragBF {
    v4u   q[2];   // two 16-byte chunks = 16 bf16
    v16bf v;
};

static __device__ __forceinline__ uint16_t f2bf_bits(float f) {
    union { __bf16 h; uint16_t u; } x;
    x.h = (__bf16)f;
    return x.u;
}

// ---------------------------------------------------------------------------
// f32 -> bf16 conversion
// ---------------------------------------------------------------------------
__global__ void cvt_f32_bf16_kernel(const float* __restrict__ in,
                                    uint16_t* __restrict__ out, long n) {
    long i = (long)blockIdx.x * blockDim.x + threadIdx.x;
    if (i < n) out[i] = f2bf_bits(in[i]);
}

// bsum = bx + bh
__global__ void bias_sum_kernel(const float* __restrict__ bx,
                                const float* __restrict__ bh,
                                float* __restrict__ bsum, int n) {
    int i = blockIdx.x * blockDim.x + threadIdx.x;
    if (i < n) bsum[i] = bx[i] + bh[i];
}

// h0 = 0 (bf16), c0 = 0 (f32)
__global__ void init_state_kernel(uint16_t* __restrict__ h_bf,
                                  float* __restrict__ c, int n) {
    int i = blockIdx.x * blockDim.x + threadIdx.x;
    if (i < n) { h_bf[i] = 0; c[i] = 0.0f; }
}

// ---------------------------------------------------------------------------
// WMMA GEMM:  C[M,N] = A[M,K](bf16,row-major) * Bw[N,K](bf16,row-major)^T
//             + add[row*addStride + col]         (f32 addend, stride 0 = bias)
//
// One wave computes a (16*MT) x (16*NT) output tile. K loop runs in blocks of
// 32 with explicit double buffering: fragments for block k+32 are loaded
// before the WMMAs on block k are issued, so the hardware wait for the
// current fragments leaves the next block's loads in flight.
// Requires K % 64 == 0.
// ---------------------------------------------------------------------------
template <int MT, int NT>
__global__ __launch_bounds__(256)
void wmma_gemm_bf16_pipelined(float* __restrict__ C,
                              const uint16_t* __restrict__ A,
                              const uint16_t* __restrict__ Bw,
                              const float* __restrict__ add,
                              long addStride,
                              int M, int N, int K) {
    const int wave   = blockIdx.x * (blockDim.x >> 5) + (threadIdx.x >> 5);
    const int lane   = threadIdx.x & 31;
    const int ngroup = N / (16 * NT);
    const int mtile  = wave / ngroup;
    const int ng     = wave % ngroup;
    const int m0     = mtile * 16 * MT;
    const int n0     = ng * 16 * NT;
    if (m0 >= M) return;

    const int mrow  = lane & 15;   // A: M row / B: N row / C: column
    const int khalf = lane >> 4;   // which K-half this lane covers

    v8f acc[MT][NT];
#pragma unroll
    for (int s = 0; s < MT; ++s)
#pragma unroll
        for (int t = 0; t < NT; ++t) acc[s][t] = (v8f){0,0,0,0,0,0,0,0};

    const uint16_t* Arow[MT];
#pragma unroll
    for (int s = 0; s < MT; ++s)
        Arow[s] = A + (long)(m0 + s * 16 + mrow) * K;
    const uint16_t* Brow[NT];
#pragma unroll
    for (int t = 0; t < NT; ++t)
        Brow[t] = Bw + (long)(n0 + t * 16 + mrow) * K;

    // A fragment (16x32): VGPR0-3 -> K = khalf*8 + 0..7,
    //                     VGPR4-7 -> K = 16 + khalf*8 + 0..7
    // B fragment (32x16): lane holds column n, same K order as A
    auto loadFrags = [&](FragBF (&a)[MT], FragBF (&b)[NT], int kb) {
#pragma unroll
        for (int s = 0; s < MT; ++s) {
            a[s].q[0] = *(const v4u*)(Arow[s] + kb + khalf * 8);
            a[s].q[1] = *(const v4u*)(Arow[s] + kb + 16 + khalf * 8);
        }
#pragma unroll
        for (int t = 0; t < NT; ++t) {
            b[t].q[0] = *(const v4u*)(Brow[t] + kb + khalf * 8);
            b[t].q[1] = *(const v4u*)(Brow[t] + kb + 16 + khalf * 8);
        }
    };
    auto doWMMA = [&](FragBF (&a)[MT], FragBF (&b)[NT]) {
#pragma unroll
        for (int s = 0; s < MT; ++s)
#pragma unroll
            for (int t = 0; t < NT; ++t)
                acc[s][t] = __builtin_amdgcn_wmma_f32_16x16x32_bf16(
                    false, a[s].v, false, b[t].v, (short)0, acc[s][t],
                    false, false);
    };

    FragBF aA[MT], bA[NT], aB[MT], bB[NT];
    loadFrags(aA, bA, 0);
    int kb = 0;
    for (; kb + 64 < K; kb += 64) {
        loadFrags(aB, bB, kb + 32);   // prefetch block k+32
        doWMMA(aA, bA);               // compute block k
        loadFrags(aA, bA, kb + 64);   // prefetch block k+64
        doWMMA(aB, bB);               // compute block k+32
    }
    // tail (kb == K-64 when K >= 64 and K % 64 == 0)
    loadFrags(aB, bB, kb + 32);
    doWMMA(aA, bA);
    doWMMA(aB, bB);

    // Epilogue: C/D layout -> row = r + 8*khalf, col = mrow
#pragma unroll
    for (int s = 0; s < MT; ++s) {
#pragma unroll
        for (int t = 0; t < NT; ++t) {
            const int col = n0 + t * 16 + mrow;
#pragma unroll
            for (int r = 0; r < 8; ++r) {
                const int row = m0 + s * 16 + r + 8 * khalf;
                float v = acc[s][t][r] + add[(long)row * addStride + col];
                C[(long)row * N + col] = v;
            }
        }
    }
}

// ---------------------------------------------------------------------------
// LSTM cell element-wise update for one timestep.
// gates layout: (B, 4H) f32, order i | f | o | g  (jnp.split order).
// ---------------------------------------------------------------------------
static __device__ __forceinline__ float sigmoidf_fast(float x) {
    return 1.0f / (1.0f + __expf(-x));
}

__global__ void lstm_cell_kernel(const float* __restrict__ gates,
                                 float* __restrict__ c,
                                 uint16_t* __restrict__ h_bf,
                                 float* __restrict__ out_t, // d_out + t*H
                                 int H, int S, int total) {
    int idx = blockIdx.x * blockDim.x + threadIdx.x;
    if (idx >= total) return;
    int b = idx / H;
    int j = idx - b * H;
    const float* g = gates + (long)b * 4 * H;

    float ig = sigmoidf_fast(g[j]);
    float fg = sigmoidf_fast(g[H + j]);
    float og = sigmoidf_fast(g[2 * H + j]);
    float gg = tanhf(g[3 * H + j]);

    float cn = fg * c[idx] + ig * gg;
    float hn = og * tanhf(cn);

    c[idx]    = cn;
    h_bf[idx] = f2bf_bits(hn);
    out_t[(long)b * S * H + j] = hn;
}

// Copy h_last (= outputs[:, S-1, :]) and c_last into the output tail.
__global__ void finalize_kernel(float* __restrict__ out,
                                const float* __restrict__ c,
                                int H, int S, int total /*B*H*/) {
    int idx = blockIdx.x * blockDim.x + threadIdx.x;
    if (idx >= total) return;
    int b = idx / H;
    int j = idx - b * H;
    long BSH = (long)total * S;   // B*S*H  (total = B*H)
    out[BSH + idx]         = out[(long)b * S * H + (long)(S - 1) * H + j];
    out[BSH + total + idx] = c[idx];
}

// ---------------------------------------------------------------------------
// Host side
// ---------------------------------------------------------------------------
extern "C" void kernel_launch(void* const* d_in, const int* in_sizes, int n_in,
                              void* d_out, int out_size, void* d_ws, size_t ws_size,
                              hipStream_t stream) {
    constexpr int B = 64, S = 512, D = 1024, H = 1024;
    constexpr int G = 4 * H;                 // 4096 gate width
    constexpr long BS = (long)B * S;         // 32768 rows of the big GEMM

    const float* x  = (const float*)d_in[0];  // (B,S,D)
    const float* Wx = (const float*)d_in[1];  // (4H,D)
    const float* bx = (const float*)d_in[2];  // (4H)
    const float* Wh = (const float*)d_in[3];  // (4H,H)
    const float* bh = (const float*)d_in[4];  // (4H)
    float* out = (float*)d_out;               // outputs (B,S,H) | h_last | c_last

    // ---- workspace carve-up (bytes, 256-aligned) ----
    char* ws = (char*)d_ws;
    size_t off = 0;
    auto alloc = [&](size_t bytes) {
        char* p = ws + off;
        off += (bytes + 255) & ~(size_t)255;
        return p;
    };
    float*    xg    = (float*)   alloc(BS * G * sizeof(float));        // 512 MB
    uint16_t* x_bf  = (uint16_t*)alloc(BS * D * sizeof(uint16_t));     //  64 MB
    uint16_t* Wx_bf = (uint16_t*)alloc((long)G * D * sizeof(uint16_t));//   8 MB
    uint16_t* Wh_bf = (uint16_t*)alloc((long)G * H * sizeof(uint16_t));//   8 MB
    float*    gates = (float*)   alloc((long)B * G * sizeof(float));   //   1 MB
    float*    cbuf  = (float*)   alloc((long)B * H * sizeof(float));
    uint16_t* h_bf  = (uint16_t*)alloc((long)B * H * sizeof(uint16_t));
    float*    bsum  = (float*)   alloc((long)G * sizeof(float));
    (void)ws_size;

    const int TB = 256;

    // ---- conversions & init ----
    {
        long n = BS * D;
        cvt_f32_bf16_kernel<<<(int)((n + TB - 1) / TB), TB, 0, stream>>>(x, x_bf, n);
    }
    {
        long n = (long)G * D;
        cvt_f32_bf16_kernel<<<(int)((n + TB - 1) / TB), TB, 0, stream>>>(Wx, Wx_bf, n);
    }
    {
        long n = (long)G * H;
        cvt_f32_bf16_kernel<<<(int)((n + TB - 1) / TB), TB, 0, stream>>>(Wh, Wh_bf, n);
    }
    bias_sum_kernel<<<(G + TB - 1) / TB, TB, 0, stream>>>(bx, bh, bsum, G);
    init_state_kernel<<<(B * H + TB - 1) / TB, TB, 0, stream>>>(h_bf, cbuf, B * H);

    // ---- big input projection: xg = x_bf @ Wx_bf^T + (bx+bh) ----
    {
        constexpr int MT = 2, NT = 2;                 // 32x32 tile per wave
        int waves  = (int)(BS / (16 * MT)) * (G / (16 * NT)); // 1024 * 128
        int blocks = waves / 8;                       // 8 waves / 256-thread block
        wmma_gemm_bf16_pipelined<MT, NT><<<blocks, TB, 0, stream>>>(
            xg, x_bf, Wx_bf, bsum, /*addStride=*/0, (int)BS, G, D);
    }

    // ---- recurrence: 512 sequential (GEMM + cell) steps ----
    for (int t = 0; t < S; ++t) {
        constexpr int MT = 1, NT = 2;                 // 16x32 tile -> 512 waves
        int waves  = (B / (16 * MT)) * (G / (16 * NT)); // 4 * 128 = 512
        int blocks = waves / 8;                       // 64 blocks
        // gates = xg[:, t, :] + h @ Wh^T   (addend row stride = S*4H)
        wmma_gemm_bf16_pipelined<MT, NT><<<blocks, TB, 0, stream>>>(
            gates, h_bf, Wh_bf, xg + (long)t * G, /*addStride=*/(long)S * G,
            B, G, H);

        lstm_cell_kernel<<<(B * H + TB - 1) / TB, TB, 0, stream>>>(
            gates, cbuf, h_bf, out + (long)t * H, H, S, B * H);
    }

    // ---- h_last / c_last tail ----
    finalize_kernel<<<(B * H + TB - 1) / TB, TB, 0, stream>>>(out, cbuf, H, S, B * H);
    (void)out_size; (void)in_sizes; (void)n_in;
}
